// MyModel_82978768159519
// MI455X (gfx1250) — matrix operand order repaired and checked by reading, not compiled, not run on hardware
//
#include <hip/hip_runtime.h>
#include <hip/hip_bf16.h>

// ---- model constants (from reference) ----
#define B_  128
#define T_  128
#define V_  10000
#define E_  256
#define H_  1024
#define D_  128
#define G3_ (3 * H_)   // 3072

typedef __attribute__((ext_vector_type(16))) __bf16 v16bf;
typedef __attribute__((ext_vector_type(8)))  __bf16 v8bf;
typedef __attribute__((ext_vector_type(8)))  float  v8f;
typedef __attribute__((ext_vector_type(4)))  unsigned v4u;
typedef __attribute__((ext_vector_type(8)))  int      v8i;
typedef __attribute__((ext_vector_type(4)))  int      v4i;

#if defined(__AMDGCN__) && __has_builtin(__builtin_amdgcn_tensor_load_to_lds)
#define USE_TDM 1
#else
#define USE_TDM 0
#endif

// f32 -> bf16 round-to-nearest-even
static __device__ __forceinline__ __bf16 f2bf(float f) {
    unsigned u = __builtin_bit_cast(unsigned, f);
    unsigned r = (u + 0x7FFFu + ((u >> 16) & 1u)) >> 16;
    unsigned short s = (unsigned short)r;
    return __builtin_bit_cast(__bf16, s);
}

// load A fragment: 16 contiguous elements p[0..15]
static __device__ __forceinline__ v16bf load_frag_a(const __bf16* p) {
    v8bf lo = __builtin_bit_cast(v8bf, *(const v4u*)p);
    v8bf hi = __builtin_bit_cast(v8bf, *(const v4u*)(p + 8));
    return __builtin_shufflevector(lo, hi, 0,1,2,3,4,5,6,7,8,9,10,11,12,13,14,15);
}
// load B fragment: 16 contiguous elements (pre-swizzled fragment-major layout)
static __device__ __forceinline__ v16bf load_frag_b(const __bf16* p) {
    v8bf lo = __builtin_bit_cast(v8bf, *(const v4u*)p);
    v8bf hi = __builtin_bit_cast(v8bf, *(const v4u*)(p + 8));
    return __builtin_shufflevector(lo, hi, 0,1,2,3,4,5,6,7,8,9,10,11,12,13,14,15);
}

// ---------------------------------------------------------------------------
// Repack + convert B[K,N] (f32, row-major) into WMMA fragment-major bf16:
//   Bp[((nt*(K/32) + kb)*32 + lane)*16 + e] = B[kb*32 + (lane>>4)*16 + e, nt*16 + (lane&15)]
// so every lane's 16 B-operand elements are 32 contiguous bytes.
// ---------------------------------------------------------------------------
__global__ void k_repack_b_f32(const float* __restrict__ Bm,
                               __bf16* __restrict__ Bp, int N, int K) {
    int idx = blockIdx.x * 256 + threadIdx.x;
    if (idx >= N * K) return;
    int e    = idx & 15;
    int lane = (idx >> 4) & 31;
    int frag = idx >> 9;
    int kblocks = K >> 5;
    int nt = frag / kblocks;
    int kb = frag - nt * kblocks;
    int n = nt * 16 + (lane & 15);
    int k = kb * 32 + ((lane >> 4) << 4) + e;
    Bp[idx] = f2bf(Bm[(size_t)k * N + n]);
}

// ---------------------------------------------------------------------------
// Embedding gather + convert: x_bf16[b*T+t, e] = bf16(emb[tokens[b,t], e])
// ---------------------------------------------------------------------------
__global__ void k_embed(const int* __restrict__ tokens,
                        const float* __restrict__ emb,
                        __bf16* __restrict__ x, int total) {
    int i = blockIdx.x * 256 + threadIdx.x;
    if (i < total) {
        int e  = i & (E_ - 1);
        int bt = i >> 8;            // /E_ (E_=256)
        int tok = tokens[bt];
        x[i] = f2bf(emb[(size_t)tok * E_ + e]);
    }
}

// ---------------------------------------------------------------------------
// bf16 WMMA GEMM: out[M,N] = A[M,K] @ B[K,N] (+bias)(+relu)
// A row-major bf16; B pre-swizzled fragment-major bf16 (k_repack_b_f32).
// One 32x16 output block per wave (2 m-tiles, B fragment reused twice),
// 8 waves / 256 threads per block. M%32==0, N%16==0, K%32==0.
// ---------------------------------------------------------------------------
__global__ void k_gemm_wmma_bf16(const __bf16* __restrict__ A,
                                 const __bf16* __restrict__ Bp,
                                 const float* __restrict__ bias,
                                 float*  __restrict__ outF,
                                 __bf16* __restrict__ outBF,
                                 int M, int N, int K, int relu) {
    const int lane = threadIdx.x & 31;
    const int wave = threadIdx.x >> 5;
    const int tilesN  = N >> 4;
    const int kblocks = K >> 5;
    const int numBlk  = (M >> 5) * tilesN;
    const int blk = blockIdx.x * 8 + wave;
    if (blk >= numBlk) return;               // wave-uniform: EXEC stays all-1
    const int mb = blk / tilesN;
    const int nt = blk - mb * tilesN;

    const int am  = lane & 15;
    const int akb = (lane >> 4) << 4;        // 0 or 16
    const __bf16* a0 = A + (size_t)(mb * 32 + am) * K + akb;
    const __bf16* a1 = a0 + (size_t)16 * K;
    const __bf16* bp = Bp + (size_t)nt * kblocks * 512 + lane * 16;

    v8f acc0 = {}, acc1 = {};
    for (int kb = 0; kb < kblocks; ++kb) {
        const int k = kb * 32;
        if (kb + 1 < kblocks) {              // next K-block prefetch
            __builtin_prefetch(a0 + k + 32, 0, 0);
            __builtin_prefetch(bp + (size_t)(kb + 1) * 512, 0, 0);
        }
        v16bf bv  = load_frag_b(bp + (size_t)kb * 512);
        v16bf av0 = load_frag_a(a0 + k);
        v16bf av1 = load_frag_a(a1 + k);
        acc0 = __builtin_amdgcn_wmma_f32_16x16x32_bf16(false, av0, false, bv,
                                                       (short)0, acc0, false, false);
        acc1 = __builtin_amdgcn_wmma_f32_16x16x32_bf16(false, av1, false, bv,
                                                       (short)0, acc1, false, false);
    }

    const int cmb = (lane >> 4) << 3;        // 0 or 8
    const int cn  = lane & 15;
    const float bb = bias ? bias[nt * 16 + cn] : 0.0f;
#pragma unroll
    for (int r = 0; r < 8; ++r) {
        float v0 = acc0[r] + bb;
        float v1 = acc1[r] + bb;
        if (relu) { v0 = fmaxf(v0, 0.0f); v1 = fmaxf(v1, 0.0f); }
        size_t o0 = (size_t)(mb * 32 +      cmb + r) * N + (nt * 16 + cn);
        size_t o1 = (size_t)(mb * 32 + 16 + cmb + r) * N + (nt * 16 + cn);
        if (outF)  { outF[o0]  = v0;       outF[o1]  = v1; }
        if (outBF) { outBF[o0] = f2bf(v0); outBF[o1] = f2bf(v1); }
    }
}

// ---------------------------------------------------------------------------
// GRU gate update for one timestep t (reset_after=True):
//   z = sig(xz+rz); r = sig(xr+rr); hh = tanh(xh + r*rh); h = z*h + (1-z)*hh
// ---------------------------------------------------------------------------
__global__ void k_gru_gate(const float* __restrict__ xg,
                           const float* __restrict__ rg,
                           float*  __restrict__ h,
                           __bf16* __restrict__ h_bf,
                           __bf16* __restrict__ hs_bf,
                           int t) {
    int idx = blockIdx.x * 256 + threadIdx.x;   // < B_*H_
    int b = idx >> 10;                          // /H_ (H_=1024)
    int j = idx & (H_ - 1);
    const float* xrow = xg + (size_t)(b * T_ + t) * G3_;
    const float* rrow = rg + (size_t)b * G3_;
    float xz = xrow[j], xr = xrow[H_ + j], xh = xrow[2 * H_ + j];
    float rz = rrow[j], rr = rrow[H_ + j], rh = rrow[2 * H_ + j];
    float z  = 1.0f / (1.0f + __expf(-(xz + rz)));
    float r  = 1.0f / (1.0f + __expf(-(xr + rr)));
    float hh = tanhf(xh + r * rh);
    float hn = z * h[idx] + (1.0f - z) * hh;
    h[idx]    = hn;
    h_bf[idx] = f2bf(hn);
    hs_bf[((size_t)b * T_ + t) * H_ + j] = f2bf(hn);
}

// ---------------------------------------------------------------------------
// In-place row softmax of logits/TEMP; one block per row.
// Row is staged in LDS via the Tensor Data Mover (1D tile, 40 KB) when the
// TDM builtin is available; plain loads otherwise.
// ---------------------------------------------------------------------------
__global__ void k_softmax_rows(float* __restrict__ out, int N) {
    __shared__ float row[V_];
    __shared__ float red[256];
    const float invT = 50.0f;                   // 1/TEMP
    size_t base = (size_t)blockIdx.x * N;
    int tid = threadIdx.x;

#if USE_TDM
    if (tid < 32) {                              // wave 0 issues the TDM load
        size_t gaddr = (size_t)(const void*)(out + base);
        unsigned lds_off = (unsigned)(uintptr_t)&row[0];
        v4u g0;
        g0[0] = 1u;                                             // count=1
        g0[1] = lds_off;                                        // lds_addr
        g0[2] = (unsigned)(gaddr & 0xFFFFFFFFu);                // global_addr lo
        g0[3] = (unsigned)((gaddr >> 32) & 0x01FFFFFFu)         // global_addr hi
              | (2u << 30);                                     // type=2 (image)
        unsigned td0 = (unsigned)N, til0 = (unsigned)N, str0 = (unsigned)N;
        v8i g1;
        g1[0] = (int)(2u << 16);                                // data_size=4B
        g1[1] = (int)((td0 & 0xFFFFu) << 16);                   // tensor_dim0 lo
        g1[2] = (int)((td0 >> 16) | (1u << 16));                // td0 hi | tensor_dim1=1
        g1[3] = (int)((til0 & 0xFFFFu) << 16);                  // tile_dim0
        g1[4] = (int)1;                                         // tile_dim1=1, tile_dim2=0
        g1[5] = (int)str0;                                      // tensor_dim0_stride lo
        g1[6] = 0;
        g1[7] = 0;
        v4i z4 = {0, 0, 0, 0};
        v8i z8 = {0, 0, 0, 0, 0, 0, 0, 0};
        __builtin_amdgcn_tensor_load_to_lds(g0, g1, z4, z4, z8, 0);
        __builtin_amdgcn_s_wait_tensorcnt(0);
    }
    __syncthreads();
    float lmax = -3.4e38f;
    for (int i = tid; i < N; i += 256) lmax = fmaxf(lmax, row[i]);
#else
    float lmax = -3.4e38f;
    for (int i = tid; i < N; i += 256) {
        float v = out[base + i];
        row[i] = v;
        lmax = fmaxf(lmax, v);
    }
#endif

    red[tid] = lmax;
    __syncthreads();
    for (int s = 128; s > 0; s >>= 1) {
        if (tid < s) red[tid] = fmaxf(red[tid], red[tid + s]);
        __syncthreads();
    }
    float m = red[0];
    __syncthreads();

    float lsum = 0.0f;
    for (int i = tid; i < N; i += 256) {
        float e = __expf((row[i] - m) * invT);
        row[i] = e;
        lsum += e;
    }
    red[tid] = lsum;
    __syncthreads();
    for (int s = 128; s > 0; s >>= 1) {
        if (tid < s) red[tid] += red[tid + s];
        __syncthreads();
    }
    float inv = 1.0f / red[0];
    for (int i = tid; i < N; i += 256) out[base + i] = row[i] * inv;
}

// ---------------------------------------------------------------------------
extern "C" void kernel_launch(void* const* d_in, const int* in_sizes, int n_in,
                              void* d_out, int out_size, void* d_ws, size_t ws_size,
                              hipStream_t stream) {
    (void)in_sizes; (void)n_in; (void)out_size; (void)ws_size;
    const int*   tokens = (const int*)  d_in[0];
    const float* emb    = (const float*)d_in[1];
    const float* gru_k  = (const float*)d_in[2];
    const float* gru_rk = (const float*)d_in[3];
    const float* gru_bi = (const float*)d_in[4];
    const float* gru_br = (const float*)d_in[5];
    const float* w1     = (const float*)d_in[6];
    const float* b1     = (const float*)d_in[7];
    const float* w2     = (const float*)d_in[8];
    const float* b2     = (const float*)d_in[9];
    float* out = (float*)d_out;

    // ---- carve workspace ----
    char* ws = (char*)d_ws;
    auto carve = [&](size_t bytes) -> char* {
        char* p = ws;
        ws += (bytes + 255) & ~(size_t)255;
        return p;
    };
    const size_t MT = (size_t)B_ * T_;                                  // 16384
    float*  xg      = (float*) carve(MT * G3_ * sizeof(float));         // 201 MB
    float*  rg      = (float*) carve((size_t)B_ * G3_ * sizeof(float));
    float*  h       = (float*) carve((size_t)B_ * H_ * sizeof(float));
    __bf16* x_bf    = (__bf16*)carve(MT * E_ * 2);
    __bf16* hs_bf   = (__bf16*)carve(MT * H_ * 2);
    __bf16* d_bf    = (__bf16*)carve(MT * D_ * 2);
    __bf16* h_bf    = (__bf16*)carve((size_t)B_ * H_ * 2);
    __bf16* gruk_p  = (__bf16*)carve((size_t)E_ * G3_ * 2);
    __bf16* grurk_p = (__bf16*)carve((size_t)H_ * G3_ * 2);
    __bf16* w1_p    = (__bf16*)carve((size_t)H_ * D_ * 2);
    __bf16* w2_p    = (__bf16*)carve((size_t)D_ * V_ * 2);

    // ---- convert + swizzle B-side weights into fragment-major bf16 ----
    {
        int n;
        n = E_ * G3_;  k_repack_b_f32<<<(n + 255) / 256, 256, 0, stream>>>(gru_k,  gruk_p,  G3_, E_);
        n = H_ * G3_;  k_repack_b_f32<<<(n + 255) / 256, 256, 0, stream>>>(gru_rk, grurk_p, G3_, H_);
        n = H_ * D_;   k_repack_b_f32<<<(n + 255) / 256, 256, 0, stream>>>(w1,     w1_p,    D_,  H_);
        n = D_ * V_;   k_repack_b_f32<<<(n + 255) / 256, 256, 0, stream>>>(w2,     w2_p,    V_,  D_);
    }

    // ---- embedding gather ----
    {
        int total = B_ * T_ * E_;
        k_embed<<<total / 256, 256, 0, stream>>>(tokens, emb, x_bf, total);
    }

    // ---- xg = x @ gru_k + gru_bi   [16384 x 3072], K=256 ----
    {
        int M = B_ * T_, N = G3_, K = E_;
        int blocks = ((M >> 5) * (N >> 4) + 7) / 8;
        k_gemm_wmma_bf16<<<blocks, 256, 0, stream>>>(x_bf, gruk_p, gru_bi,
                                                     xg, nullptr, M, N, K, 0);
    }

    // ---- h0 = 0 ----
    (void)hipMemsetAsync(h,    0, (size_t)B_ * H_ * sizeof(float), stream);
    (void)hipMemsetAsync(h_bf, 0, (size_t)B_ * H_ * 2,             stream);

    // ---- sequential GRU scan over T ----
    {
        int M = B_, N = G3_, K = H_;
        int gblocks = ((M >> 5) * (N >> 4) + 7) / 8;          // 96
        int eblocks = (B_ * H_) / 256;                        // 512
        for (int t = 0; t < T_; ++t) {
            k_gemm_wmma_bf16<<<gblocks, 256, 0, stream>>>(h_bf, grurk_p, gru_br,
                                                          rg, nullptr, M, N, K, 0);
            k_gru_gate<<<eblocks, 256, 0, stream>>>(xg, rg, h, h_bf, hs_bf, t);
        }
    }

    // ---- d = relu(hs @ w1 + b1) -> bf16   [16384 x 128], K=1024 ----
    {
        int M = B_ * T_, N = D_, K = H_;
        int blocks = ((M >> 5) * (N >> 4) + 7) / 8;
        k_gemm_wmma_bf16<<<blocks, 256, 0, stream>>>(hs_bf, w1_p, b1,
                                                     nullptr, d_bf, M, N, K, 1);
    }

    // ---- logits = d @ w2 + b2 -> d_out (f32)   [16384 x 10000], K=128 ----
    {
        int M = B_ * T_, N = V_, K = D_;
        int blocks = ((M >> 5) * (N >> 4) + 7) / 8;           // 40000
        k_gemm_wmma_bf16<<<blocks, 256, 0, stream>>>(d_bf, w2_p, b2,
                                                     out, nullptr, M, N, K, 0);
    }

    // ---- softmax(logits / TEMP) in place ----
    k_softmax_rows<<<B_ * T_, 256, 0, stream>>>(out, V_);
}
